// SelfAttention_56075093016752
// MI455X (gfx1250) — matrix operand order repaired and checked by reading, not compiled, not run on hardware
//
#include <hip/hip_runtime.h>
#include <hip/hip_bf16.h>

typedef __attribute__((ext_vector_type(16))) __bf16 v16bf;
typedef __attribute__((ext_vector_type(8)))  float  v8f;

#define DMODEL 1024
#define SEQ    2048
#define BATCH  2
#define NHEADS 16
#define DK     64
#define GM     (BATCH * SEQ)   /* 4096 rows */

union BfFrag { uint4 u[2]; v16bf v; };

__device__ __forceinline__ unsigned short f2bf(float f) {
  union { float f; unsigned u; } c; c.f = f;
  unsigned u = c.u;
  u += 0x7FFFu + ((u >> 16) & 1u);   // round-to-nearest-even
  return (unsigned short)(u >> 16);
}

__device__ __forceinline__ v8f wmma_bf16(const BfFrag& a, const BfFrag& b, v8f c) {
  return __builtin_amdgcn_wmma_f32_16x16x32_bf16(false, a.v, false, b.v,
                                                 (short)0, c, false, false);
}

// CDNA5 async global->LDS copy, 16B per lane (ASYNCcnt-tracked; bypasses VGPRs).
// INST_OFFSET is added to both the LDS and global addresses (ISA 08_async_tensor §4.4).
__device__ __forceinline__ void async_g2l_32B(void* lds_dst, const void* gsrc) {
  unsigned loff = (unsigned)(uintptr_t)lds_dst;   // addr[31:0] == LDS offset
  asm volatile("global_load_async_to_lds_b128 %0, %1, off"
               :: "v"(loff), "v"(gsrc) : "memory");
  asm volatile("global_load_async_to_lds_b128 %0, %1, off offset:16"
               :: "v"(loff), "v"(gsrc) : "memory");
}
__device__ __forceinline__ void wait_async0() {
  asm volatile("s_wait_asynccnt 0x0" ::: "memory");
}

// ---------------------------------------------------------------- converts
__global__ __launch_bounds__(256)
void cvt_f32_bf16x8(const float* __restrict__ src, unsigned short* __restrict__ dst, int n8) {
  int t = blockIdx.x * 256 + threadIdx.x;
  if (t >= n8) return;
  const float4* s = (const float4*)src + (size_t)t * 2;
  float4 a = s[0], b = s[1];
  uint4 o;
  o.x = (unsigned)f2bf(a.x) | ((unsigned)f2bf(a.y) << 16);
  o.y = (unsigned)f2bf(a.z) | ((unsigned)f2bf(a.w) << 16);
  o.z = (unsigned)f2bf(b.x) | ((unsigned)f2bf(b.y) << 16);
  o.w = (unsigned)f2bf(b.z) | ((unsigned)f2bf(b.w) << 16);
  ((uint4*)dst)[t] = o;
}

// W [K][N] fp32 row-major  ->  Wt [N][K] bf16 row-major
__global__ __launch_bounds__(256)
void transp_cvt(const float* __restrict__ W, unsigned short* __restrict__ Wt) {
  __shared__ float tile[32][33];
  int n0 = blockIdx.x * 32, k0 = blockIdx.y * 32;
  int tx = threadIdx.x, ty = threadIdx.y;          // 32 x 8
#pragma unroll
  for (int i = 0; i < 32; i += 8)
    tile[ty + i][tx] = W[(size_t)(k0 + ty + i) * DMODEL + n0 + tx];
  __syncthreads();
#pragma unroll
  for (int i = 0; i < 32; i += 8)
    Wt[(size_t)(n0 + ty + i) * DMODEL + k0 + tx] = f2bf(tile[tx][ty + i]);
}

// ---------------------------------------------------------------- GEMM
// C[4096][1024] = A @ Wt^T + bias ; A,Wt bf16, accum fp32.
// Block tile 128x128, 8 waves (4M x 2N), wave tile 32x64, K-step 32.
// Double-buffered LDS fed by GLOBAL_LOAD_ASYNC_TO_LDS_B128 (ASYNCcnt),
// row pad 8 halfs (16B) to spread banks.
// MODE 0: bf16 out [b][h][s][d]   (Q, K)
// MODE 1: bf16 out [b][h][d][s]   (V transposed)
// MODE 2: fp32 out [m][n]         (final projection)
template <int MODE>
__global__ __launch_bounds__(256)
void gemm_wmma(const unsigned short* __restrict__ A,
               const unsigned short* __restrict__ Bt,
               const float* __restrict__ bias,
               void* __restrict__ outp) {
  __shared__ unsigned short As[2][128][40];
  __shared__ unsigned short Bs[2][128][40];

  const int tid  = threadIdx.x;
  const int lane = tid & 31;
  const int wave = tid >> 5;
  const int l15  = lane & 15;
  const int lh   = lane >> 4;
  const int waveM = wave >> 1, waveN = wave & 1;
  const int blockM = blockIdx.x * 128, blockN = blockIdx.y * 128;
  const int srow = tid >> 1, shalf = (tid & 1) * 16;   // staging: 2 thr/row, 32B each

  v8f acc[2][4];
#pragma unroll
  for (int mi = 0; mi < 2; ++mi)
#pragma unroll
    for (int ni = 0; ni < 4; ++ni)
#pragma unroll
      for (int i = 0; i < 8; ++i) acc[mi][ni][i] = 0.f;

  auto stage = [&](int buf, int k0) {
    async_g2l_32B(&As[buf][srow][shalf],
                  A + (size_t)(blockM + srow) * DMODEL + k0 + shalf);
    async_g2l_32B(&Bs[buf][srow][shalf],
                  Bt + (size_t)(blockN + srow) * DMODEL + k0 + shalf);
  };

  stage(0, 0);
  wait_async0();
  __syncthreads();

  for (int kk = 0; kk < DMODEL / 32; ++kk) {
    const int cur = kk & 1;
    if (kk + 1 < DMODEL / 32) stage(cur ^ 1, (kk + 1) * 32);  // overlap with WMMA

    BfFrag af[2], bf[4];
#pragma unroll
    for (int mi = 0; mi < 2; ++mi) {
      int r = waveM * 32 + mi * 16 + l15;
      af[mi].u[0] = *(const uint4*)&As[cur][r][lh * 8];
      af[mi].u[1] = *(const uint4*)&As[cur][r][16 + lh * 8];
    }
#pragma unroll
    for (int ni = 0; ni < 4; ++ni) {
      int c = waveN * 64 + ni * 16 + l15;
      bf[ni].u[0] = *(const uint4*)&Bs[cur][c][lh * 8];
      bf[ni].u[1] = *(const uint4*)&Bs[cur][c][16 + lh * 8];
    }
#pragma unroll
    for (int mi = 0; mi < 2; ++mi)
#pragma unroll
      for (int ni = 0; ni < 4; ++ni)
        acc[mi][ni] = wmma_bf16(af[mi], bf[ni], acc[mi][ni]);

    wait_async0();          // next buffer's async copies done
    __syncthreads();        // all waves done reading cur / writing nxt
  }

#pragma unroll
  for (int mi = 0; mi < 2; ++mi)
#pragma unroll
    for (int ni = 0; ni < 4; ++ni) {
      int m0 = blockM + waveM * 32 + mi * 16;
      int n  = blockN + waveN * 64 + ni * 16 + l15;
      float bv = bias[n];
#pragma unroll
      for (int i = 0; i < 8; ++i) {
        int m = m0 + i + lh * 8;
        float v = acc[mi][ni][i] + bv;
        if (MODE == 0) {
          int b = m >> 11, s = m & 2047, h = n >> 6, d = n & 63;
          ((unsigned short*)outp)[((size_t)((b * NHEADS + h) * SEQ + s)) * DK + d] = f2bf(v);
        } else if (MODE == 1) {
          int b = m >> 11, s = m & 2047, h = n >> 6, d = n & 63;
          ((unsigned short*)outp)[((size_t)((b * NHEADS + h) * DK + d)) * SEQ + s] = f2bf(v);
        } else {
          ((float*)outp)[(size_t)m * DMODEL + n] = v;
        }
      }
    }
}

// ---------------------------------------------------------------- attention
// Q,K bf16 [b][h][s][64]; Vt bf16 [b][h][64][s]; out bf16 [4096][1024].
// 8 waves/block, each wave = 16 queries, streaming-softmax over 32 keys/iter.
__global__ __launch_bounds__(256)
void attn_flash(const unsigned short* __restrict__ Q,
                const unsigned short* __restrict__ K,
                const unsigned short* __restrict__ Vt,
                unsigned short* __restrict__ O) {
  __shared__ unsigned short Pst[8][16][40];   // per-wave P re-swizzle patch

  const int tid = threadIdx.x, lane = tid & 31, wave = tid >> 5;
  const int l15 = lane & 15, lh = lane >> 4;
  const int bid = blockIdx.x;
  const int qb = bid & 15, h = (bid >> 4) & 15, b = bid >> 8;
  const size_t head = (size_t)(b * NHEADS + h) * SEQ * DK;
  const unsigned short* Qh = Q  + head;
  const unsigned short* Kh = K  + head;
  const unsigned short* Vh = Vt + head;       // [64][2048]
  const int q0 = qb * 128 + wave * 16;

  BfFrag qf[2];
#pragma unroll
  for (int f = 0; f < 2; ++f) {
    const unsigned short* p = &Qh[(size_t)(q0 + l15) * DK + f * 32 + lh * 8];
    qf[f].u[0] = *(const uint4*)p;
    qf[f].u[1] = *(const uint4*)(p + 16);
  }

  float mrow[8], lrow[8];
  v8f o[4];
#pragma unroll
  for (int i = 0; i < 8; ++i) { mrow[i] = -1e30f; lrow[i] = 0.f; }
#pragma unroll
  for (int dt = 0; dt < 4; ++dt)
#pragma unroll
    for (int i = 0; i < 8; ++i) o[dt][i] = 0.f;

  const float scale = 0.125f;  // 1/sqrt(64)

  for (int kb0 = 0; kb0 < SEQ; kb0 += 32) {
    // ---- batch-load all K fragments for this 32-key block (2 tiles x 2 d-halves)
    BfFrag kf[2][2];
#pragma unroll
    for (int kt = 0; kt < 2; ++kt) {
      const unsigned short* kp = &Kh[(size_t)(kb0 + kt * 16 + l15) * DK];
      kf[kt][0].u[0] = *(const uint4*)(kp + lh * 8);
      kf[kt][0].u[1] = *(const uint4*)(kp + 16 + lh * 8);
      kf[kt][1].u[0] = *(const uint4*)(kp + 32 + lh * 8);
      kf[kt][1].u[1] = *(const uint4*)(kp + 32 + 16 + lh * 8);
    }
    // ---- batch-load all V fragments early (consumed after softmax)
    BfFrag vf[4];
#pragma unroll
    for (int dt = 0; dt < 4; ++dt) {
      const unsigned short* vp = &Vh[(size_t)(dt * 16 + l15) * SEQ + kb0 + lh * 8];
      vf[dt].u[0] = *(const uint4*)vp;
      vf[dt].u[1] = *(const uint4*)(vp + 16);
    }
    if (kb0 + 32 < SEQ) {   // prefetch next key block (global_prefetch_b8)
      __builtin_prefetch(&Kh[(size_t)(kb0 + 32 + l15) * DK], 0, 1);
      __builtin_prefetch(&Vh[(size_t)(l15) * SEQ + kb0 + 32], 0, 1);
    }

    v8f s0, s1;
#pragma unroll
    for (int i = 0; i < 8; ++i) { s0[i] = 0.f; s1[i] = 0.f; }
    s0 = wmma_bf16(qf[0], kf[0][0], s0);
    s0 = wmma_bf16(qf[1], kf[0][1], s0);
    s1 = wmma_bf16(qf[0], kf[1][0], s1);
    s1 = wmma_bf16(qf[1], kf[1][1], s1);

    // ---- streaming softmax over these 32 keys (rows split across 16-lane groups)
#pragma unroll
    for (int i = 0; i < 8; ++i) {
      float x0 = s0[i] * scale, x1 = s1[i] * scale;
      float mx = fmaxf(x0, x1);
#pragma unroll
      for (int m = 1; m < 16; m <<= 1) mx = fmaxf(mx, __shfl_xor(mx, m, 16));
      float mnew = fmaxf(mrow[i], mx);
      float c  = __expf(mrow[i] - mnew);
      float p0 = __expf(x0 - mnew);
      float p1 = __expf(x1 - mnew);
      float ps = p0 + p1;
#pragma unroll
      for (int m = 1; m < 16; m <<= 1) ps += __shfl_xor(ps, m, 16);
      lrow[i] = lrow[i] * c + ps;
      mrow[i] = mnew;
#pragma unroll
      for (int dt = 0; dt < 4; ++dt) o[dt][i] *= c;
      int qr = i + lh * 8;
      Pst[wave][qr][l15]      = f2bf(p0);
      Pst[wave][qr][16 + l15] = f2bf(p1);
    }
    // ---- reload P in A-operand layout, then P @ V
    BfFrag pf;
    pf.u[0] = *(const uint4*)&Pst[wave][l15][lh * 8];
    pf.u[1] = *(const uint4*)&Pst[wave][l15][16 + lh * 8];
#pragma unroll
    for (int dt = 0; dt < 4; ++dt)
      o[dt] = wmma_bf16(pf, vf[dt], o[dt]);
  }

#pragma unroll
  for (int i = 0; i < 8; ++i) {
    float inv = 1.0f / lrow[i];
    int m = b * SEQ + q0 + i + lh * 8;
#pragma unroll
    for (int dt = 0; dt < 4; ++dt) {
      int n = h * DK + dt * 16 + l15;
      O[(size_t)m * DMODEL + n] = f2bf(o[dt][i] * inv);
    }
  }
}

// ---------------------------------------------------------------- launch
extern "C" void kernel_launch(void* const* d_in, const int* in_sizes, int n_in,
                              void* d_out, int out_size, void* d_ws, size_t ws_size,
                              hipStream_t stream) {
  (void)in_sizes; (void)n_in; (void)out_size; (void)ws_size;
  const float* x  = (const float*)d_in[0];
  const float* wq = (const float*)d_in[1];
  const float* bq = (const float*)d_in[2];
  const float* wk = (const float*)d_in[3];
  const float* bk = (const float*)d_in[4];
  const float* wv = (const float*)d_in[5];
  const float* bv = (const float*)d_in[6];
  const float* wo = (const float*)d_in[7];
  const float* bo = (const float*)d_in[8];

  char* ws = (char*)d_ws;
  unsigned short* xb   = (unsigned short*)(ws);                        // 8 MB
  unsigned short* wqt  = (unsigned short*)(ws + ( 8u << 20));          // 2 MB each
  unsigned short* wkt  = (unsigned short*)(ws + (10u << 20));
  unsigned short* wvt  = (unsigned short*)(ws + (12u << 20));
  unsigned short* wot  = (unsigned short*)(ws + (14u << 20));
  unsigned short* qbuf = (unsigned short*)(ws + (16u << 20));          // 8 MB
  unsigned short* kbuf = (unsigned short*)(ws + (24u << 20));          // 8 MB
  unsigned short* vtb  = (unsigned short*)(ws + (32u << 20));          // 8 MB
  unsigned short* attn = (unsigned short*)(ws + (40u << 20));          // 8 MB

  // 1) convert inputs to bf16
  cvt_f32_bf16x8<<<(GM * DMODEL / 8 + 255) / 256, 256, 0, stream>>>(x, xb, GM * DMODEL / 8);
  dim3 tg(32, 32), tb(32, 8);
  transp_cvt<<<tg, tb, 0, stream>>>(wq, wqt);
  transp_cvt<<<tg, tb, 0, stream>>>(wk, wkt);
  transp_cvt<<<tg, tb, 0, stream>>>(wv, wvt);
  transp_cvt<<<tg, tb, 0, stream>>>(wo, wot);

  // 2) Q/K/V projections (WMMA + async LDS staging)
  dim3 gg(GM / 128, DMODEL / 128);
  gemm_wmma<0><<<gg, 256, 0, stream>>>(xb, wqt, bq, qbuf);
  gemm_wmma<0><<<gg, 256, 0, stream>>>(xb, wkt, bk, kbuf);
  gemm_wmma<1><<<gg, 256, 0, stream>>>(xb, wvt, bv, vtb);

  // 3) flash attention (WMMA)
  attn_flash<<<BATCH * NHEADS * (SEQ / 128), 256, 0, stream>>>(qbuf, kbuf, vtb, attn);

  // 4) output projection -> fp32 d_out
  gemm_wmma<2><<<gg, 256, 0, stream>>>(attn, wot, bo, (float*)d_out);
}
